// Mamba_54614804136613
// MI455X (gfx1250) — compile-verified, matrix operand order checked by reading, not run on hardware
//
#include <hip/hip_runtime.h>
#include <hip/hip_bf16.h>

#define D_MODEL 1024
#define D_INNER 2048
#define D_STATE 16
#define D_CONV  4
#define DT_RANK 64
#define BATCH   2
#define SEQLEN  4096
#define NTOK    (BATCH*SEQLEN)              // 8192 tokens
#define NPROJ   (DT_RANK + 2*D_STATE)       // 96

typedef __bf16 bf16_t;
typedef __attribute__((ext_vector_type(16))) __bf16 v16bf;
typedef __attribute__((ext_vector_type(8)))  __bf16 v8bf;
typedef __attribute__((ext_vector_type(8)))  float  v8f;

// ---------------------------------------------------------------- fp32->bf16
__global__ void k_cvt_bf16(const float* __restrict__ s, bf16_t* __restrict__ d, int n) {
  int i = blockIdx.x * blockDim.x + threadIdx.x;
  if (i < n) d[i] = (bf16_t)s[i];
}

// --------------------------------------------------- register-blocked WMMA
// One wave computes an (MT*16) x (NT*16) output block:
//   per 32-wide K step: MT+NT fragment loads, MT*NT v_wmma issues.
// Both operands row-major along K (A row m contiguous in k; W row n
// contiguous in k). Per ISA 16-bit A/B layout: lanes 0-15 own K
// {0..7,16..23}, lanes 16-31 own K {8..15,24..31} of each K step.
template <int MT, int NT>
__device__ __forceinline__ void wmma_block(const bf16_t* __restrict__ A, int lda,
                                           const bf16_t* __restrict__ W, int ldw,
                                           int K, int m0, int n0, int lane,
                                           v8f acc[MT][NT]) {
  const v8f zacc = {};
#pragma unroll
  for (int i = 0; i < MT; ++i)
#pragma unroll
    for (int j = 0; j < NT; ++j) acc[i][j] = zacc;

  const int koff = (lane & 16) ? 8 : 0;
  const bf16_t* pa[MT];
  const bf16_t* pw[NT];
#pragma unroll
  for (int i = 0; i < MT; ++i) pa[i] = A + (m0 + i * 16 + (lane & 15)) * lda + koff;
#pragma unroll
  for (int j = 0; j < NT; ++j) pw[j] = W + (n0 + j * 16 + (lane & 15)) * ldw + koff;

  for (int kb = 0; kb < K; kb += 32) {
    __builtin_prefetch(pa[0] + kb + 64, 0, 1);
    __builtin_prefetch(pw[0] + kb + 64, 0, 1);
    v16bf a[MT], b[NT];
#pragma unroll
    for (int i = 0; i < MT; ++i) {
      v8bf lo = *(const v8bf*)(pa[i] + kb);
      v8bf hi = *(const v8bf*)(pa[i] + kb + 16);
#pragma unroll
      for (int q = 0; q < 8; ++q) { a[i][q] = lo[q]; a[i][q + 8] = hi[q]; }
    }
#pragma unroll
    for (int j = 0; j < NT; ++j) {
      v8bf lo = *(const v8bf*)(pw[j] + kb);
      v8bf hi = *(const v8bf*)(pw[j] + kb + 16);
#pragma unroll
      for (int q = 0; q < 8; ++q) { b[j][q] = lo[q]; b[j][q + 8] = hi[q]; }
    }
#pragma unroll
    for (int i = 0; i < MT; ++i)
#pragma unroll
      for (int j = 0; j < NT; ++j)
        acc[i][j] = __builtin_amdgcn_wmma_f32_16x16x32_bf16(
            false, a[i], false, b[j], (short)0, acc[i][j], false, false);
  }
}

// --------------------------------------------- GEMM1: xz = hs @ W_in^T
// split interleaved columns (e even -> x channel d=e/2, e odd -> z) into
// (b,d,l) layout, l contiguous.
__global__ void k_gemm_xz(const bf16_t* __restrict__ A, const bf16_t* __restrict__ W,
                          float* __restrict__ xbuf, float* __restrict__ zbuf) {
  const int lane = threadIdx.x & 31;
  const int wave = threadIdx.x >> 5;
  const int m0 = blockIdx.x * 64;
  const int n0 = (blockIdx.y * 2 + wave) * 64;
  v8f acc[4][4];
  wmma_block<4, 4>(A, D_MODEL, W, D_MODEL, D_MODEL, m0, n0, lane, acc);
#pragma unroll
  for (int j = 0; j < 4; ++j) {
    const int e = n0 + j * 16 + (lane & 15);
    float* dst = (e & 1) ? zbuf : xbuf;
    const int d = e >> 1;
#pragma unroll
    for (int i = 0; i < 4; ++i)
#pragma unroll
      for (int r = 0; r < 8; ++r) {
        int m = m0 + i * 16 + r + ((lane & 16) ? 8 : 0);
        int b = m >> 12, l = m & (SEQLEN - 1);
        dst[(b * D_INNER + d) * SEQLEN + l] = acc[i][j][r];
      }
  }
}

// ------------------------------- causal depthwise conv(4) + bias + SiLU
__global__ void k_conv_silu(const float* __restrict__ xbuf,
                            const float* __restrict__ conv_w,
                            const float* __restrict__ conv_b,
                            float* __restrict__ xconv,
                            bf16_t* __restrict__ xcbf) {
  int t = blockIdx.x * blockDim.x + threadIdx.x;   // BATCH*D_INNER*(SEQLEN/32)
  int chunk = t & ((SEQLEN / 32) - 1);
  int bd = t >> 7;
  int d = bd & (D_INNER - 1);
  int b = bd >> 11;
  int l0 = chunk * 32;
  const float* xr = xbuf + (size_t)bd * SEQLEN;
  float w0 = conv_w[d*4+0], w1 = conv_w[d*4+1], w2 = conv_w[d*4+2], w3 = conv_w[d*4+3];
  float bias = conv_b[d];
  float p3 = (l0 >= 3) ? xr[l0-3] : 0.f;
  float p2 = (l0 >= 2) ? xr[l0-2] : 0.f;
  float p1 = (l0 >= 1) ? xr[l0-1] : 0.f;
  for (int i = 0; i < 32; ++i) {
    int l = l0 + i;
    float xv = xr[l];
    float acc = bias + w0*p3 + w1*p2 + w2*p1 + w3*xv;
    p3 = p2; p2 = p1; p1 = xv;
    float sv = acc / (1.f + __expf(-acc));           // SiLU
    xconv[(size_t)bd * SEQLEN + l] = sv;
    xcbf[(size_t)(b * SEQLEN + l) * D_INNER + d] = (bf16_t)sv;  // (b,l,d) for GEMM2
  }
}

// --------------------------------------------- GEMM2: x_dbl = x @ x_proj_w^T
__global__ void k_gemm_xdbl(const bf16_t* __restrict__ A, const bf16_t* __restrict__ W,
                            float* __restrict__ xdbl, bf16_t* __restrict__ dtbf) {
  const int lane = threadIdx.x & 31;
  const int wave = threadIdx.x >> 5;
  const int m0 = blockIdx.x * 64;
  const int n0 = (blockIdx.y * 2 + wave) * 48;   // NT=3 -> 48 cols / wave
  v8f acc[4][3];
  wmma_block<4, 3>(A, D_INNER, W, D_INNER, D_INNER, m0, n0, lane, acc);
#pragma unroll
  for (int j = 0; j < 3; ++j) {
    const int n = n0 + j * 16 + (lane & 15);
#pragma unroll
    for (int i = 0; i < 4; ++i)
#pragma unroll
      for (int r = 0; r < 8; ++r) {
        int m = m0 + i * 16 + r + ((lane & 16) ? 8 : 0);
        float v = acc[i][j][r];
        xdbl[m * NPROJ + n] = v;
        if (n < DT_RANK) dtbf[m * DT_RANK + n] = (bf16_t)v;
      }
  }
}

// ------------------------- GEMM3: delta = softplus(dt @ dt_w^T + dt_b[d])
__global__ void k_gemm_delta(const bf16_t* __restrict__ A, const bf16_t* __restrict__ W,
                             const float* __restrict__ dt_b, float* __restrict__ delta) {
  const int lane = threadIdx.x & 31;
  const int wave = threadIdx.x >> 5;
  const int m0 = blockIdx.x * 64;
  const int n0 = (blockIdx.y * 2 + wave) * 64;
  v8f acc[4][4];
  wmma_block<4, 4>(A, DT_RANK, W, DT_RANK, DT_RANK, m0, n0, lane, acc);
#pragma unroll
  for (int j = 0; j < 4; ++j) {
    const int d = n0 + j * 16 + (lane & 15);
    float bias = dt_b[d];
#pragma unroll
    for (int i = 0; i < 4; ++i)
#pragma unroll
      for (int r = 0; r < 8; ++r) {
        int m = m0 + i * 16 + r + ((lane & 16) ? 8 : 0);
        float v = acc[i][j][r] + bias;
        v = (v > 20.f) ? v : log1pf(__expf(v));      // softplus
        int b = m >> 12, l = m & (SEQLEN - 1);
        delta[(b * D_INNER + d) * SEQLEN + l] = v;
      }
  }
}

// --------------------------------- selective scan: 1 thread per (b,d) channel
// B/C (shared by all channels of a batch) staged through LDS in 64-step chunks.
#define LCHUNK 64
__global__ void k_scan(const float* __restrict__ delta, const float* __restrict__ xconv,
                       const float* __restrict__ xdbl, const float* __restrict__ zbuf,
                       const float* __restrict__ A_log, const float* __restrict__ Dvec,
                       bf16_t* __restrict__ ybf) {
  __shared__ float sBC[LCHUNK][32];                // [l][0..15]=B, [16..31]=C
  int t = blockIdx.x * blockDim.x + threadIdx.x;   // BATCH*D_INNER total
  int d = t & (D_INNER - 1);
  int b = t >> 11;                                 // constant across the block
  float Areg[D_STATE];
#pragma unroll
  for (int n = 0; n < D_STATE; ++n) Areg[n] = -__expf(A_log[d * D_STATE + n]);
  float Dv = Dvec[d];
  const float* dr  = delta + (size_t)t * SEQLEN;
  const float* xr  = xconv + (size_t)t * SEQLEN;
  const float* zr  = zbuf  + (size_t)t * SEQLEN;
  const float* dbl = xdbl  + (size_t)b * SEQLEN * NPROJ;
  float s[D_STATE];
#pragma unroll
  for (int n = 0; n < D_STATE; ++n) s[n] = 0.f;

  for (int l0 = 0; l0 < SEQLEN; l0 += LCHUNK) {
    __syncthreads();
    // cooperative load: LCHUNK*32 floats, 16 per thread, coalesced over c
#pragma unroll
    for (int k = 0; k < (LCHUNK * 32) / 128; ++k) {
      int idx = threadIdx.x + k * 128;
      int ll = idx >> 5, c = idx & 31;
      sBC[ll][c] = dbl[(l0 + ll) * NPROJ + DT_RANK + c];
    }
    __syncthreads();

    for (int li = 0; li < LCHUNK; ++li) {
      int l = l0 + li;
      float dt = dr[l];
      float xv = xr[l];
      float dx = dt * xv;
      float y = 0.f;
      const float* bc = &sBC[li][0];
#pragma unroll
      for (int n = 0; n < D_STATE; ++n) {
        float w = __expf(dt * Areg[n]);
        s[n] = fmaf(s[n], w, dx * bc[n]);
        y = fmaf(s[n], bc[16 + n], y);
      }
      y = fmaf(Dv, xv, y);                          // + D*x
      float zv = zr[l];
      y *= zv / (1.f + __expf(-zv));                // * silu(z)
      ybf[(size_t)(b * SEQLEN + l) * D_INNER + d] = (bf16_t)y;   // (b,l,d)
    }
  }
}

// --------------------------------------------- GEMM4: out = y @ W_out^T
__global__ void k_gemm_out(const bf16_t* __restrict__ A, const bf16_t* __restrict__ W,
                           float* __restrict__ out) {
  const int lane = threadIdx.x & 31;
  const int wave = threadIdx.x >> 5;
  const int m0 = blockIdx.x * 64;
  const int n0 = (blockIdx.y * 2 + wave) * 64;
  v8f acc[4][4];
  wmma_block<4, 4>(A, D_INNER, W, D_INNER, D_INNER, m0, n0, lane, acc);
#pragma unroll
  for (int j = 0; j < 4; ++j) {
    const int n = n0 + j * 16 + (lane & 15);
#pragma unroll
    for (int i = 0; i < 4; ++i)
#pragma unroll
      for (int r = 0; r < 8; ++r) {
        int m = m0 + i * 16 + r + ((lane & 16) ? 8 : 0);
        out[m * D_MODEL + n] = acc[i][j][r];
      }
  }
}

// ----------------------------------------------------------------- launcher
extern "C" void kernel_launch(void* const* d_in, const int* in_sizes, int n_in,
                              void* d_out, int out_size, void* d_ws, size_t ws_size,
                              hipStream_t stream) {
  (void)in_sizes; (void)n_in; (void)out_size; (void)ws_size;
  const float* hs     = (const float*)d_in[0];
  const float* W_in   = (const float*)d_in[1];
  const float* conv_w = (const float*)d_in[2];
  const float* conv_b = (const float*)d_in[3];
  const float* x_proj = (const float*)d_in[4];
  const float* dt_w   = (const float*)d_in[5];
  const float* dt_b   = (const float*)d_in[6];
  const float* A_log  = (const float*)d_in[7];
  const float* Dvec   = (const float*)d_in[8];
  const float* W_out  = (const float*)d_in[9];
  float* out = (float*)d_out;

  char* ws = (char*)d_ws;
  size_t off = 0;
  auto alloc = [&](size_t bytes) -> void* {
    void* p = ws + off;
    off += (bytes + 255) & ~((size_t)255);
    return p;
  };
  bf16_t* hs_bf    = (bf16_t*)alloc((size_t)NTOK * D_MODEL * 2);
  bf16_t* win_bf   = (bf16_t*)alloc((size_t)2 * D_INNER * D_MODEL * 2);
  bf16_t* wout_bf  = (bf16_t*)alloc((size_t)D_MODEL * D_INNER * 2);
  bf16_t* xproj_bf = (bf16_t*)alloc((size_t)NPROJ * D_INNER * 2);
  bf16_t* dtw_bf   = (bf16_t*)alloc((size_t)D_INNER * DT_RANK * 2);
  float*  xbuf     = (float*) alloc((size_t)BATCH * D_INNER * SEQLEN * 4);
  float*  zbuf     = (float*) alloc((size_t)BATCH * D_INNER * SEQLEN * 4);
  float*  xconv    = (float*) alloc((size_t)BATCH * D_INNER * SEQLEN * 4);
  bf16_t* xcbf     = (bf16_t*)alloc((size_t)NTOK * D_INNER * 2);
  float*  xdbl     = (float*) alloc((size_t)NTOK * NPROJ * 4);
  bf16_t* dtbf     = (bf16_t*)alloc((size_t)NTOK * DT_RANK * 2);
  float*  delta    = (float*) alloc((size_t)BATCH * D_INNER * SEQLEN * 4);
  bf16_t* ybf      = (bf16_t*)alloc((size_t)NTOK * D_INNER * 2);

  auto cvt = [&](const float* s, bf16_t* dst, int n) {
    k_cvt_bf16<<<(n + 255) / 256, 256, 0, stream>>>(s, dst, n);
  };
  cvt(hs,     hs_bf,    NTOK * D_MODEL);
  cvt(W_in,   win_bf,   2 * D_INNER * D_MODEL);
  cvt(x_proj, xproj_bf, NPROJ * D_INNER);
  cvt(dt_w,   dtw_bf,   D_INNER * DT_RANK);
  cvt(W_out,  wout_bf,  D_MODEL * D_INNER);

  dim3 blk(64);  // 2 waves per block, one 64-wide (or 48-wide) N strip each
  k_gemm_xz   <<<dim3(NTOK / 64, (2 * D_INNER) / 128), blk, 0, stream>>>(hs_bf, win_bf, xbuf, zbuf);
  k_conv_silu <<<(BATCH * D_INNER * (SEQLEN / 32)) / 256, 256, 0, stream>>>(xbuf, conv_w, conv_b, xconv, xcbf);
  k_gemm_xdbl <<<dim3(NTOK / 64, 1), blk, 0, stream>>>(xcbf, xproj_bf, xdbl, dtbf);
  k_gemm_delta<<<dim3(NTOK / 64, D_INNER / 128), blk, 0, stream>>>(dtbf, dtw_bf, dt_b, delta);
  k_scan      <<<(BATCH * D_INNER) / 128, 128, 0, stream>>>(delta, xconv, xdbl, zbuf, A_log, Dvec, ybf);
  k_gemm_out  <<<dim3(NTOK / 64, D_MODEL / 128), blk, 0, stream>>>(ybf, wout_bf, out);
}
